// GAT_71365176590742
// MI455X (gfx1250) — compile-verified, matrix operand order checked by reading, not run on hardware
//
#include <hip/hip_runtime.h>
#include <hip/hip_bf16.h>

typedef __attribute__((ext_vector_type(16))) _Float16 v16h;
typedef __attribute__((ext_vector_type(8)))  float    v8f;

#define NEG_SLOPE 0.2f
#define FLT_BIG 3.402823466e+38f

// ---------------------------------------------------------------------------
// float atomic max via signed/unsigned integer ordering trick
// ---------------------------------------------------------------------------
__device__ __forceinline__ void atomicMaxF32(float* addr, float v) {
    if (v >= 0.0f) atomicMax((int*)addr, __float_as_int(v));
    else           atomicMin((unsigned int*)addr, __float_as_uint(v));
}

__device__ __forceinline__ float leaky(float v) {
    return v >= 0.0f ? v : NEG_SLOPE * v;
}

// ---------------------------------------------------------------------------
// Pack W [K x ncols] (f32, row-major) into WMMA-B fragment-major hi/lo f16.
// Fragment f = (tileN * NK + tileK) * 32 + lane holds 16 halfs:
//   col = tileN*16 + (lane&15), kb = (lane>>4)*8,
//   j<8 -> K = tileK*32 + kb + j ; j>=8 -> K = tileK*32 + kb + 8 + j
// Out-of-range cols pad with zero, so the GEMM inner loop needs no guards.
// ---------------------------------------------------------------------------
__global__ void pack_B_kernel(const float* __restrict__ W, int ldb, int ncols,
                              int K, _Float16* __restrict__ hi,
                              _Float16* __restrict__ lo) {
    const int nk = K >> 5;
    const int ntN = (ncols + 15) >> 4;
    const int total = ntN * nk * 32 * 16;
    const int t = blockIdx.x * blockDim.x + threadIdx.x;
    if (t >= total) return;
    const int j    = t & 15;
    const int lane = (t >> 4) & 31;
    const int tile = t >> 9;
    const int tk = tile % nk;
    const int tn = tile / nk;
    const int col = tn * 16 + (lane & 15);
    const int kb  = (lane >> 4) << 3;
    const int k   = tk * 32 + kb + (j < 8 ? j : j + 8);
    const float f = (col < ncols) ? W[(size_t)k * ldb + col] : 0.0f;
    const _Float16 h = (_Float16)f;
    hi[t] = h;
    lo[t] = (_Float16)(f - (float)h);
}

// ---------------------------------------------------------------------------
// WMMA GEMM: C = A[MxK](f32) * B[KxNcols], B pre-packed hi/lo f16 fragments.
// hi*hi + hi*lo + lo*hi => ~fp32 accuracy at f16 WMMA rate.
// One wave computes one 16x16 tile; K templated so the loop fully unrolls.
// Rows assumed multiple of 16 (N = 50000 = 3125*16).
// ---------------------------------------------------------------------------
template <int K>
__global__ void wmma_gemm_hilo(const float* __restrict__ A, int lda,
                               const _Float16* __restrict__ Bhi,
                               const _Float16* __restrict__ Blo, int ncols,
                               float* __restrict__ C, int ldc) {
    constexpr int NK = K / 32;
    const int wave  = threadIdx.x >> 5;
    const int tileN = blockIdx.y * (blockDim.x >> 5) + wave;
    const int c0 = tileN * 16;
    if (c0 >= ((ncols + 15) & ~15)) return;   // wave-uniform: EXEC stays full

    const int lane = threadIdx.x & 31;
    const int lr = lane & 15;
    const int kb = (lane >> 4) << 3;
    const int r0 = blockIdx.x * 16;

    const v16h* bh = (const v16h*)Bhi + (size_t)tileN * NK * 32 + lane;
    const v16h* bl = (const v16h*)Blo + (size_t)tileN * NK * 32 + lane;

    v8f acc = {};
#pragma unroll
    for (int tk = 0; tk < NK; ++tk) {
        // A fragment (16x32): row = r0+lr, K = tk*32+kb+{0..7} and +{16..23}
        const float* pa = A + (size_t)(r0 + lr) * lda + tk * 32 + kb;
        float fa[16];
#pragma unroll
        for (int j = 0; j < 8; ++j) { fa[j] = pa[j]; fa[8 + j] = pa[16 + j]; }
        v16h ahi, alo;
#pragma unroll
        for (int j = 0; j < 16; ++j) {
            _Float16 h = (_Float16)fa[j];
            ahi[j] = h;
            alo[j] = (_Float16)(fa[j] - (float)h);
        }
        const v16h bhi_f = bh[(size_t)tk * 32];
        const v16h blo_f = bl[(size_t)tk * 32];

        acc = __builtin_amdgcn_wmma_f32_16x16x32_f16(false, ahi, false, bhi_f,
                                                     (short)0, acc, false, false);
        acc = __builtin_amdgcn_wmma_f32_16x16x32_f16(false, ahi, false, blo_f,
                                                     (short)0, acc, false, false);
        acc = __builtin_amdgcn_wmma_f32_16x16x32_f16(false, alo, false, bhi_f,
                                                     (short)0, acc, false, false);
    }
    // store C: VGPR v -> M = v + 8*(lane>=16), N = lane&15
    const int mh = (lane >> 4) << 3;
#pragma unroll
    for (int v = 0; v < 8; ++v) {
        const int row = r0 + v + mh;
        const int col = c0 + lr;
        if (col < ncols) C[(size_t)row * ldc + col] = acc[v];
    }
}

// ---------------------------------------------------------------------------
// el[n,h] = sum_d h[n,h,d]*al[h,d] ; er likewise
// ---------------------------------------------------------------------------
__global__ void node_dots(const float* __restrict__ h, const float* __restrict__ al,
                          const float* __restrict__ ar, float* __restrict__ el,
                          float* __restrict__ er, int N, int H, int D) {
    const int t = blockIdx.x * blockDim.x + threadIdx.x;
    if (t >= N * H) return;
    const int n = t / H, hd = t % H;
    const float* p = h + (size_t)n * H * D + (size_t)hd * D;
    const float* a = al + hd * D;
    const float* b = ar + hd * D;
    float sl = 0.0f, sr = 0.0f;
    for (int i = 0; i < D; ++i) { float v = p[i]; sl += v * a[i]; sr += v * b[i]; }
    el[t] = sl; er[t] = sr;
}

// ---------------------------------------------------------------------------
// zero accumulator, set max=-inf, denom=0
// ---------------------------------------------------------------------------
__global__ void init_layer(float* __restrict__ mx, float* __restrict__ dn,
                           float* __restrict__ acc, int nh, long total) {
    const long t = (long)blockIdx.x * blockDim.x + threadIdx.x;
    if (t < total) acc[t] = 0.0f;
    if (t < nh) { mx[t] = -FLT_BIG; dn[t] = 0.0f; }
}

// ---------------------------------------------------------------------------
// edge passes: logit max, exp-sum (denominator), weighted scatter
// ---------------------------------------------------------------------------
__global__ void edge_max_kernel(const int* __restrict__ src, const int* __restrict__ dst,
                                const float* __restrict__ el, const float* __restrict__ er,
                                float* __restrict__ mx, int E, int H) {
    const long t = (long)blockIdx.x * blockDim.x + threadIdx.x;
    if (t >= (long)E * H) return;
    const int e = (int)(t / H), hd = (int)(t % H);
    const int s = src[e], d = dst[e];
    const float v = leaky(el[s * H + hd] + er[d * H + hd]);
    atomicMaxF32(&mx[d * H + hd], v);
}

__global__ void edge_den_kernel(const int* __restrict__ src, const int* __restrict__ dst,
                                const float* __restrict__ el, const float* __restrict__ er,
                                const float* __restrict__ mx, float* __restrict__ dn,
                                int E, int H) {
    const long t = (long)blockIdx.x * blockDim.x + threadIdx.x;
    if (t >= (long)E * H) return;
    const int e = (int)(t / H), hd = (int)(t % H);
    const int s = src[e], d = dst[e];
    const float v = leaky(el[s * H + hd] + er[d * H + hd]);
    atomicAdd(&dn[d * H + hd], expf(v - mx[d * H + hd]));
}

// one block per edge, one thread per output channel c in [0, H*D)
__global__ void edge_scatter_kernel(const int* __restrict__ src, const int* __restrict__ dst,
                                    const float* __restrict__ el, const float* __restrict__ er,
                                    const float* __restrict__ mx, const float* __restrict__ dn,
                                    const float* __restrict__ h, float* __restrict__ acc,
                                    int H, int D) {
    const int e = blockIdx.x;
    const int c = threadIdx.x;
    const int HD = H * D;
    if (c >= HD) return;
    const int s = src[e], d = dst[e];
    const int hd = c / D;
    const float v = leaky(el[s * H + hd] + er[d * H + hd]);
    const float alpha = expf(v - mx[d * H + hd]) / fmaxf(dn[d * H + hd], 1e-9f);
    atomicAdd(&acc[(size_t)d * HD + c], h[(size_t)s * HD + c] * alpha);
}

// out = acc + bias [+ ELU]
__global__ void bias_act_kernel(float* __restrict__ io, const float* __restrict__ b,
                                long total, int HD, int elu) {
    const long t = (long)blockIdx.x * blockDim.x + threadIdx.x;
    if (t >= total) return;
    float v = io[t] + b[t % HD];
    if (elu) v = v > 0.0f ? v : expm1f(v);
    io[t] = v;
}

// ---------------------------------------------------------------------------
extern "C" void kernel_launch(void* const* d_in, const int* in_sizes, int n_in,
                              void* d_out, int out_size, void* d_ws, size_t ws_size,
                              hipStream_t stream) {
    const float* x   = (const float*)d_in[0];
    const int*   src = (const int*)d_in[1];
    const int*   dst = (const int*)d_in[2];
    const float* W1  = (const float*)d_in[3];
    const float* al1 = (const float*)d_in[4];
    const float* ar1 = (const float*)d_in[5];
    const float* b1  = (const float*)d_in[6];
    const float* W2  = (const float*)d_in[7];
    const float* al2 = (const float*)d_in[8];
    const float* ar2 = (const float*)d_in[9];
    const float* b2  = (const float*)d_in[10];

    const int IN_F = 128, H1 = 8, D1 = 32, HD1 = H1 * D1, NCLS = 40;
    const int N = in_sizes[0] / IN_F;
    const int E = in_sizes[1];

    float* ws = (float*)d_ws;
    size_t o = 0;
    float* h1  = ws + o; o += (size_t)N * HD1;   // layer-1 pre-attention features
    float* a1  = ws + o; o += (size_t)N * HD1;   // layer-1 accumulator / layer-2 input
    float* el1 = ws + o; o += (size_t)N * H1;
    float* er1 = ws + o; o += (size_t)N * H1;
    float* mx1 = ws + o; o += (size_t)N * H1;
    float* dn1 = ws + o; o += (size_t)N * H1;
    // packed weights (hi/lo f16 fragments)
    const int NP1 = (HD1 / 16) * (IN_F / 32) * 32 * 16;         // 32768 halfs
    const int NP2 = ((NCLS + 15) / 16) * (HD1 / 32) * 32 * 16;  // 12288 halfs
    _Float16* w1hi = (_Float16*)(ws + o);
    _Float16* w1lo = w1hi + NP1;
    _Float16* w2hi = w1lo + NP1;
    _Float16* w2lo = w2hi + NP2;
    o += (size_t)(2 * NP1 + 2 * NP2 + 1) / 2;    // halfs -> floats
    // layer-2 temporaries reuse h1's space (free after layer-1 scatter)
    float* h2  = h1;                              // N * NCLS
    float* el2 = h1 + (size_t)N * NCLS;           // N
    float* er2 = el2 + N;
    float* mx2 = er2 + N;
    float* dn2 = mx2 + N;
    float* out = (float*)d_out;

    const long tot1 = (long)N * HD1;
    const long tot2 = (long)N * NCLS;
    const long eh1  = (long)E * H1;

    // ---------------- weight packing (tiny, once per launch) ----------------
    pack_B_kernel<<<(NP1 + 255) / 256, 256, 0, stream>>>(W1, HD1, HD1, IN_F,
                                                         w1hi, w1lo);
    pack_B_kernel<<<(NP2 + 255) / 256, 256, 0, stream>>>(W2, NCLS, NCLS, HD1,
                                                         w2hi, w2lo);

    // ---------------- layer 1 ----------------
    wmma_gemm_hilo<128><<<dim3(N / 16, 2), 256, 0, stream>>>(x, IN_F, w1hi, w1lo,
                                                             HD1, h1, HD1);
    node_dots<<<(N * H1 + 255) / 256, 256, 0, stream>>>(h1, al1, ar1, el1, er1,
                                                        N, H1, D1);
    init_layer<<<(int)((tot1 + 255) / 256), 256, 0, stream>>>(mx1, dn1, a1,
                                                              N * H1, tot1);
    edge_max_kernel<<<(int)((eh1 + 255) / 256), 256, 0, stream>>>(src, dst, el1, er1,
                                                                  mx1, E, H1);
    edge_den_kernel<<<(int)((eh1 + 255) / 256), 256, 0, stream>>>(src, dst, el1, er1,
                                                                  mx1, dn1, E, H1);
    edge_scatter_kernel<<<E, HD1, 0, stream>>>(src, dst, el1, er1, mx1, dn1,
                                               h1, a1, H1, D1);
    bias_act_kernel<<<(int)((tot1 + 255) / 256), 256, 0, stream>>>(a1, b1, tot1,
                                                                   HD1, 1);

    // ---------------- layer 2 ----------------
    wmma_gemm_hilo<256><<<dim3(N / 16, 1), 96, 0, stream>>>(a1, HD1, w2hi, w2lo,
                                                            NCLS, h2, NCLS);
    node_dots<<<(N + 255) / 256, 256, 0, stream>>>(h2, al2, ar2, el2, er2,
                                                   N, 1, NCLS);
    init_layer<<<(int)((tot2 + 255) / 256), 256, 0, stream>>>(mx2, dn2, out,
                                                              N, tot2);
    edge_max_kernel<<<(E + 255) / 256, 256, 0, stream>>>(src, dst, el2, er2,
                                                         mx2, E, 1);
    edge_den_kernel<<<(E + 255) / 256, 256, 0, stream>>>(src, dst, el2, er2,
                                                         mx2, dn2, E, 1);
    edge_scatter_kernel<<<E, 64, 0, stream>>>(src, dst, el2, er2, mx2, dn2,
                                              h2, out, 1, NCLS);
    bias_act_kernel<<<(int)((tot2 + 255) / 256), 256, 0, stream>>>(out, b2, tot2,
                                                                   NCLS, 0);
}
